// WindowedMaskedAttentionCalculation_82712480186753
// MI455X (gfx1250) — matrix operand order repaired and checked by reading, not compiled, not run on hardware
//
#include <hip/hip_runtime.h>

// Flash-attention for B=8,H=8,S=2048,D=64 fp32 tensors on gfx1250 (wave32).
// f16 WMMA (v_wmma_f32_16x16x32_f16) with f32 accumulate, online softmax with
// v_permlane16 butterfly reductions, async global->LDS staging of
// pos_scores/mask (ASYNCcnt), transposed V tile in LDS so every matrix
// fragment load is a ds_load_b128.

typedef __attribute__((ext_vector_type(16))) _Float16 v16h;
typedef __attribute__((ext_vector_type(8)))  _Float16 v8h;
typedef __attribute__((ext_vector_type(8)))  float    v8f;
typedef __attribute__((ext_vector_type(4)))  float    v4f;
typedef __attribute__((ext_vector_type(4)))  int      v4i;

#define S_LEN  2048
#define HEADS  8
#define DHEAD  64
#define KBLK   32                 // keys per iteration
#define NKB    (S_LEN / KBLK)     // 64 iterations
#define QROWS  128                // query rows per block (8 waves x 16)
#define INV_T  0.125f
#define PM_W   (KBLK + 4)         // pos/mask LDS row (floats), bank-skewed

#if __has_builtin(__builtin_amdgcn_global_load_async_to_lds_b128)
#define HAVE_ASYNC_LDS 1
#else
#define HAVE_ASYNC_LDS 0
#endif

// Exact pointer types the async-LDS builtin expects:
// arg0: int4* in global (AS1), arg1: int4* in LDS (AS3), then imm offset, cpol.
typedef __attribute__((address_space(1))) v4i gbl_v4i;
typedef __attribute__((address_space(3))) v4i lds_v4i;

// Copy a 128-row x 32-float tile (row stride = grow_stride floats in global)
// into LDS [128][PM_W]. Async (ASYNCcnt) when available, sync fallback else.
__device__ __forceinline__ void stage_pm_tile(const float* __restrict__ gsrc,
                                              size_t grow_stride,
                                              float* lds_base, int tid)
{
    #pragma unroll
    for (int j = 0; j < 4; ++j) {
        const int c   = tid + 256 * j;     // 1024 chunks of 16B
        const int row = c >> 3;
        const int col = (c & 7) * 4;
        const float* g = gsrc + (size_t)row * grow_stride + col;
        float*       l = lds_base + row * PM_W + col;
#if HAVE_ASYNC_LDS
        __builtin_amdgcn_global_load_async_to_lds_b128(
            (gbl_v4i*)g, (lds_v4i*)l, 0, 0);
#else
        *(v4f*)l = *(const v4f*)g;
#endif
    }
}

__device__ __forceinline__ void wait_async_all()
{
#if HAVE_ASYNC_LDS
#if __has_builtin(__builtin_amdgcn_s_wait_asynccnt)
    __builtin_amdgcn_s_wait_asynccnt(0);
#else
    asm volatile("s_wait_asynccnt 0x0" ::: "memory");
#endif
#endif
}

// xor-butterfly within 16-lane rows via v_permlane16_b32 (VALU, keeps the LDS
// pipe free for WMMA fragment loads). sel tables encode lane -> lane^m.
__device__ __forceinline__ float perm16_xor(float x, unsigned lo, unsigned hi, int m)
{
#if __has_builtin(__builtin_amdgcn_permlane16)
    int i = __float_as_int(x);
    i = __builtin_amdgcn_permlane16(i, i, lo, hi, false, false);
    return __int_as_float(i);
#else
    return __shfl_xor(x, m, 32);
#endif
}
__device__ __forceinline__ float red_max16(float x)
{
    x = fmaxf(x, perm16_xor(x, 0x67452301u, 0xEFCDAB89u, 1));
    x = fmaxf(x, perm16_xor(x, 0x54761032u, 0xDCFE98BAu, 2));
    x = fmaxf(x, perm16_xor(x, 0x32107654u, 0xBA98FEDCu, 4));
    x = fmaxf(x, perm16_xor(x, 0xFEDCBA98u, 0x76543210u, 8));
    return x;
}
__device__ __forceinline__ float red_sum16(float x)
{
    x += perm16_xor(x, 0x67452301u, 0xEFCDAB89u, 1);
    x += perm16_xor(x, 0x54761032u, 0xDCFE98BAu, 2);
    x += perm16_xor(x, 0x32107654u, 0xBA98FEDCu, 4);
    x += perm16_xor(x, 0xFEDCBA98u, 0x76543210u, 8);
    return x;
}

__global__ __launch_bounds__(256)
void fa_wmma_kernel(const float* __restrict__ q,
                    const float* __restrict__ k,
                    const float* __restrict__ v,
                    const float* __restrict__ pos,
                    const float* __restrict__ mk,
                    float* __restrict__ out)
{
    __shared__ __align__(16) _Float16 Klds [KBLK ][DHEAD + 8];   // row-major keys
    __shared__ __align__(16) _Float16 VldsT[DHEAD][KBLK  + 8];   // V transposed [d][kk]
    __shared__ __align__(16) _Float16 Plds [8][16][KBLK  + 8];   // wave-private P bounce
    __shared__ __align__(16) float    PosLds [2][QROWS][PM_W];   // double-buffered
    __shared__ __align__(16) float    MaskLds[2][QROWS][PM_W];

    const int bh   = blockIdx.x;          // b*H + h
    const int h    = bh & (HEADS - 1);
    const int qb   = blockIdx.y;
    const int tid  = threadIdx.x;
    const int wave = tid >> 5;
    const int lane = tid & 31;
    const int hi   = lane >> 4;           // 16-lane half
    const int l16  = lane & 15;

    const float* qbase = q + (size_t)bh * S_LEN * DHEAD;
    const float* kbase = k + (size_t)bh * S_LEN * DHEAD;
    const float* vbase = v + (size_t)bh * S_LEN * DHEAD;
    const float* pbase = pos + (size_t)h * S_LEN * S_LEN
                             + (size_t)(qb * QROWS) * S_LEN;     // pos tile rows
    const float* mbase = mk  + (size_t)(qb * QROWS) * S_LEN;     // mask tile rows

    const int qrow0 = qb * QROWS + wave * 16;

    // ---- Q A-fragments (16x32 f16 x2). contiguous k order used here: lanes 0-15 row=l16 with
    // K=0..15; lanes 16-31 row=l16 with K=16..31. ----
    v16h aQ0, aQ1;
    {
        const float* qr = qbase + (size_t)(qrow0 + l16) * DHEAD;
        const int kb0 = hi * 16;
        #pragma unroll
        for (int i = 0; i < 16; ++i) {
            const int kk = kb0 + i;
            aQ0[i] = (_Float16)qr[kk];
            aQ1[i] = (_Float16)qr[32 + kk];
        }
    }

    // ---- Accumulators ----
    v8f o[4];
    #pragma unroll
    for (int t = 0; t < 4; ++t)
        #pragma unroll
        for (int r = 0; r < 8; ++r) o[t][r] = 0.0f;

    float m_run[8], l_run[8];
    #pragma unroll
    for (int r = 0; r < 8; ++r) { m_run[r] = -1.0e30f; l_run[r] = 0.0f; }

    // Prologue: launch async staging of pos/mask for key block 0.
    stage_pm_tile(pbase, S_LEN, &PosLds [0][0][0], tid);
    stage_pm_tile(mbase, S_LEN, &MaskLds[0][0][0], tid);

    // ==================== stream over key blocks ====================
    for (int kb = 0; kb < NKB; ++kb) {
        const int buf = kb & 1;

        wait_async_all();      // pos/mask[buf] transfers done (this wave)
        __syncthreads();       // visible block-wide; K/V LDS free for reuse

        // ---- Stage K (row-major) and V (transposed) as f16 ----
        {
            // K tile: thread -> (row = tid/8, 8 consecutive d)
            const int row = tid >> 3;
            const int c0  = (tid & 7) * 8;
            const float* kg = kbase + (size_t)(kb * KBLK + row) * DHEAD + c0;
            const v4f ka = *(const v4f*)kg;
            const v4f kc = *(const v4f*)(kg + 4);
            v8h kh;
            #pragma unroll
            for (int i = 0; i < 4; ++i) { kh[i] = (_Float16)ka[i]; kh[4 + i] = (_Float16)kc[i]; }
            *(v8h*)&Klds[row][c0] = kh;

            // V tile transposed: thread -> (d = tid%64, 8 consecutive kk)
            const int d   = tid & 63;
            const int kk0 = (tid >> 6) * 8;
            const float* vg = vbase + (size_t)(kb * KBLK + kk0) * DHEAD + d;
            v8h vh;
            #pragma unroll
            for (int i = 0; i < 8; ++i) vh[i] = (_Float16)vg[(size_t)i * DHEAD];
            *(v8h*)&VldsT[d][kk0] = vh;

            if (kb + 1 < NKB) {                    // global_prefetch_b8
                __builtin_prefetch(kg + KBLK * DHEAD, 0, 1);
                __builtin_prefetch(vg + KBLK * DHEAD, 0, 1);
            }
        }

        // Kick off async pos/mask staging for the NEXT key block (other buffer).
        if (kb + 1 < NKB) {
            const size_t col = (size_t)(kb + 1) * KBLK;
            stage_pm_tile(pbase + col, S_LEN, &PosLds [buf ^ 1][0][0], tid);
            stage_pm_tile(mbase + col, S_LEN, &MaskLds[buf ^ 1][0][0], tid);
        }

        __syncthreads();       // K/V tiles visible

        // ---- QK^T: 16x32 scores, 2 n-tiles x 2 K-chunk WMMAs ----
        v8f s[2];
        #pragma unroll
        for (int nt = 0; nt < 2; ++nt) {
            #pragma unroll
            for (int r = 0; r < 8; ++r) s[nt][r] = 0.0f;
            const int key = nt * 16 + l16;     // B column = key within block
            const int kr0 = hi * 16;           // B k order used here (contiguous): lanes<16 K=0..15, else 16..31
            const v8h b0lo = *(const v8h*)&Klds[key][kr0];
            const v8h b0hi = *(const v8h*)&Klds[key][kr0 + 8];
            const v8h b1lo = *(const v8h*)&Klds[key][32 + kr0];
            const v8h b1hi = *(const v8h*)&Klds[key][32 + kr0 + 8];
            v16h b0, b1;
            #pragma unroll
            for (int i = 0; i < 8; ++i) {
                b0[i] = b0lo[i]; b0[8 + i] = b0hi[i];
                b1[i] = b1lo[i]; b1[8 + i] = b1hi[i];
            }
            s[nt] = __builtin_amdgcn_wmma_f32_16x16x32_f16(
                        false, aQ0, false, b0, (short)0, s[nt], false, false);
            s[nt] = __builtin_amdgcn_wmma_f32_16x16x32_f16(
                        false, aQ1, false, b1, (short)0, s[nt], false, false);
        }

        // ---- scale + pos + mask + online softmax (C layout: lane=key col,
        //      VGPR r = query row hi*8+r). pos/mask come from LDS buffers. ----
        #pragma unroll
        for (int r = 0; r < 8; ++r) {
            const int rl = wave * 16 + hi * 8 + r;   // row within 128-row tile
            const float e0 = MaskLds[buf][rl][l16]      * (s[0][r] * INV_T + PosLds[buf][rl][l16]);
            const float e1 = MaskLds[buf][rl][16 + l16] * (s[1][r] * INV_T + PosLds[buf][rl][16 + l16]);

            const float mb   = red_max16(fmaxf(e0, e1));
            const float mnew = fmaxf(m_run[r], mb);
            const float corr = __expf(m_run[r] - mnew);
            const float p0 = __expf(e0 - mnew);
            const float p1 = __expf(e1 - mnew);
            const float rs = red_sum16(p0 + p1);

            l_run[r] = l_run[r] * corr + rs;
            m_run[r] = mnew;
            #pragma unroll
            for (int t = 0; t < 4; ++t) o[t][r] *= corr;

            Plds[wave][hi * 8 + r][l16]      = (_Float16)p0;
            Plds[wave][hi * 8 + r][16 + l16] = (_Float16)p1;
        }

        // ---- P x V: A-fragment from Plds (2x b128), B from VldsT (2x b128) ----
        v16h aP;
        {
            const int kb0 = hi * 8;   // runs [kb0,kb0+8) and [kb0+16,kb0+24)
            const v8h plo = *(const v8h*)&Plds[wave][l16][kb0];
            const v8h phi = *(const v8h*)&Plds[wave][l16][kb0 + 16];
            #pragma unroll
            for (int i = 0; i < 8; ++i) { aP[i] = plo[i]; aP[8 + i] = phi[i]; }
        }
        #pragma unroll
        for (int t = 0; t < 4; ++t) {
            const int dcol = t * 16 + l16;
            const int kk0  = hi * 8;
            const v8h vlo = *(const v8h*)&VldsT[dcol][kk0];
            const v8h vhi = *(const v8h*)&VldsT[dcol][kk0 + 16];
            v16h bV;
            #pragma unroll
            for (int i = 0; i < 8; ++i) { bV[i] = vlo[i]; bV[8 + i] = vhi[i]; }
            o[t] = __builtin_amdgcn_wmma_f32_16x16x32_f16(
                       false, aP, false, bV, (short)0, o[t], false, false);
        }
    }

    // ---- Normalize (reciprocal once per row) and store, coalesced ----
    float inv_l[8];
    #pragma unroll
    for (int r = 0; r < 8; ++r) inv_l[r] = 1.0f / l_run[r];

    float* obase = out + (size_t)bh * S_LEN * DHEAD;
    #pragma unroll
    for (int t = 0; t < 4; ++t) {
        const int dcol = t * 16 + l16;
        #pragma unroll
        for (int r = 0; r < 8; ++r) {
            const int mg = qrow0 + hi * 8 + r;
            obase[(size_t)mg * DHEAD + dcol] = o[t][r] * inv_l[r];
        }
    }
}

extern "C" void kernel_launch(void* const* d_in, const int* in_sizes, int n_in,
                              void* d_out, int out_size, void* d_ws, size_t ws_size,
                              hipStream_t stream) {
    const float* q   = (const float*)d_in[0];
    const float* k   = (const float*)d_in[1];
    const float* v   = (const float*)d_in[2];
    const float* pos = (const float*)d_in[3];
    const float* mk  = (const float*)d_in[4];
    float* out = (float*)d_out;

    const int BH = in_sizes[0] / (S_LEN * DHEAD);   // B*H = 64
    dim3 grid(BH, S_LEN / QROWS);                   // (64, 16)
    fa_wmma_kernel<<<grid, 256, 0, stream>>>(q, k, v, pos, mk, out);
}
